// Propagation_4612794876549
// MI455X (gfx1250) — compile-verified, matrix operand order checked by reading, not run on hardware
//
#include <hip/hip_runtime.h>

// ---------------------------------------------------------------------------
// Problem constants (match reference)
// ---------------------------------------------------------------------------
#define Bc   2
#define DIMc 256
#define Hc   4
#define DHc  64
#define Nc   4096
#define DIM2 512

typedef _Float16 v16h __attribute__((ext_vector_type(16)));
typedef float    v8f  __attribute__((ext_vector_type(8)));
typedef unsigned int uintx4 __attribute__((ext_vector_type(4)));
typedef int intx4 __attribute__((ext_vector_type(4)));
typedef int intx8 __attribute__((ext_vector_type(8)));

__device__ inline _Float16 us2h(unsigned short u) {
  union { unsigned short u; _Float16 h; } c; c.u = u; return c.h;
}

__device__ inline float red_max16(float v) {
#pragma unroll
  for (int m = 1; m < 16; m <<= 1) v = fmaxf(v, __shfl_xor(v, m, 32));
  return v;
}
__device__ inline float red_sum16(float v) {
#pragma unroll
  for (int m = 1; m < 16; m <<= 1) v += __shfl_xor(v, m, 32);
  return v;
}

// ---------------------------------------------------------------------------
// Tensor Data Mover: 2D tile (elem = 2 bytes) global -> LDS.
// D# layout per CDNA5 ISA ch.8:
//  group0: [1:0]=count=1, [63:32]=lds_addr, [120:64]=global_addr, [127:126]=2
//  group1: [17:16]=data_size(1=2B), [79:48]=tensor_dim0, [111:80]=tensor_dim1,
//          [127:112]=tile_dim0, [143:128]=tile_dim1, [207:160]=dim0_stride
// ---------------------------------------------------------------------------
__device__ inline void tdm_load_2d_f16(unsigned lds_off, const void* gaddr,
                                       unsigned tile_x, unsigned tile_y,
                                       unsigned tensor_x, unsigned tensor_y,
                                       unsigned stride_x) {
  unsigned long long ga = (unsigned long long)(uintptr_t)gaddr;
  uintx4 g0;
  g0[0] = 1u;                                    // count=1, user descriptor
  g0[1] = lds_off;                               // lds_addr (bytes)
  g0[2] = (unsigned)(ga & 0xffffffffu);          // global_addr[95:64]
  g0[3] = (unsigned)((ga >> 32) & 0x1ffffffu) | (2u << 30);  // addr hi | type=2
  intx8 g1;
  g1[0] = (int)(1u << 16);                       // data_size = 1 -> 2 bytes
  g1[1] = (int)((tensor_x & 0xffffu) << 16);     // tensor_dim0[15:0]
  g1[2] = (int)(((tensor_x >> 16) & 0xffffu) | ((tensor_y & 0xffffu) << 16));
  g1[3] = (int)(((tensor_y >> 16) & 0xffffu) | ((tile_x & 0xffffu) << 16));
  g1[4] = (int)(tile_y & 0xffffu);               // tile_dim1 (tile_dim2=0)
  g1[5] = (int)stride_x;                         // dim0_stride[31:0]
  g1[6] = 0;                                     // dim0_stride hi | dim1_stride lo
  g1[7] = 0;
  intx4 z4 = {0, 0, 0, 0};
#if defined(__clang_major__) && (__clang_major__ >= 23)
  intx8 z8 = {0, 0, 0, 0, 0, 0, 0, 0};
  __builtin_amdgcn_tensor_load_to_lds(g0, g1, z4, z4, z8, 0);
#else
  __builtin_amdgcn_tensor_load_to_lds(g0, g1, z4, z4, 0);
#endif
}

// ---------------------------------------------------------------------------
// Shared GEMM helpers: block = 256 threads (8 waves), tile 64(O) x 128(N),
// K-step 32.  Staging is two-phase (batch all global loads, then cvt+store)
// so multiple loads stay in flight.  LDS rows are K-contiguous so each WMMA
// fragment is two 16B (ds_load_b128) loads per lane; fragments are batch-
// loaded before the dependent WMMA chain so LDS fetch overlaps matrix math.
//   ldsW : [64 o][32 k] halves      ldsX : [128 n][32 k] halves
// ---------------------------------------------------------------------------
__device__ inline void stage_w_tile(const float* __restrict__ W, int ostride,
                                    int k0, int o0, int t, _Float16* ldsW) {
  float4 wv[2];
#pragma unroll
  for (int i = 0; i < 2; ++i) {
    int lin = t + i * 256;
    wv[i] = *(const float4*)&W[(size_t)(k0 + (lin >> 4)) * ostride + o0 +
                               (lin & 15) * 4];
  }
#pragma unroll
  for (int i = 0; i < 2; ++i) {
    int lin = t + i * 256;
    int kk = lin >> 4, og = (lin & 15) * 4;
    ldsW[(og + 0) * 32 + kk] = (_Float16)wv[i].x;
    ldsW[(og + 1) * 32 + kk] = (_Float16)wv[i].y;
    ldsW[(og + 2) * 32 + kk] = (_Float16)wv[i].z;
    ldsW[(og + 3) * 32 + kk] = (_Float16)wv[i].w;
  }
}

__device__ inline void stage_x_f32(const float* __restrict__ X, int k0, int n0,
                                   int t, _Float16* ldsX) {
  float4 xv[4];
#pragma unroll
  for (int i = 0; i < 4; ++i) {
    int lin = t + i * 256;
    xv[i] = *(const float4*)&X[(size_t)(k0 + (lin >> 5)) * Nc + n0 +
                               (lin & 31) * 4];
  }
#pragma unroll
  for (int i = 0; i < 4; ++i) {
    int lin = t + i * 256;
    int kk = lin >> 5, ng = (lin & 31) * 4;
    ldsX[(ng + 0) * 32 + kk] = (_Float16)xv[i].x;
    ldsX[(ng + 1) * 32 + kk] = (_Float16)xv[i].y;
    ldsX[(ng + 2) * 32 + kk] = (_Float16)xv[i].z;
    ldsX[(ng + 3) * 32 + kk] = (_Float16)xv[i].w;
  }
}

__device__ inline void stage_x_f16(const _Float16* __restrict__ X, int k0,
                                   int n0, int t, _Float16* ldsX) {
  ushort4 raw[4];
#pragma unroll
  for (int i = 0; i < 4; ++i) {
    int lin = t + i * 256;
    raw[i] = *(const ushort4*)&X[(size_t)(k0 + (lin >> 5)) * Nc + n0 +
                                 (lin & 31) * 4];
  }
#pragma unroll
  for (int i = 0; i < 4; ++i) {
    int lin = t + i * 256;
    int kk = lin >> 5, ng = (lin & 31) * 4;
    ldsX[(ng + 0) * 32 + kk] = us2h(raw[i].x);
    ldsX[(ng + 1) * 32 + kk] = us2h(raw[i].y);
    ldsX[(ng + 2) * 32 + kk] = us2h(raw[i].z);
    ldsX[(ng + 3) * 32 + kk] = us2h(raw[i].w);
  }
}

__device__ inline void wmma_tile_step(const _Float16* ldsW,
                                      const _Float16* ldsX, int osub, int nsub,
                                      int L, v8f acc[4]) {
  v16h a = *(const v16h*)&ldsW[(osub + (L & 15)) * 32 + ((L >> 4) * 16)];
  v16h bf[4];
#pragma unroll
  for (int f = 0; f < 4; ++f)
    bf[f] =
        *(const v16h*)&ldsX[(nsub + f * 16 + (L & 15)) * 32 + ((L >> 4) * 16)];
#pragma unroll
  for (int f = 0; f < 4; ++f)
    acc[f] = __builtin_amdgcn_wmma_f32_16x16x32_f16(false, a, false, bf[f],
                                                    (short)0, acc[f], false,
                                                    false);
}

// ---------------------------------------------------------------------------
// Kernel 1: QKV projection.  Out[o,n] = sum_i W[i,o] * X[b,i,n] + bias[o]
// vmode 0: scatter to [b,h,pos,d]   (Q and K layout, d contiguous)
// vmode 1: scatter to [b,h,d,pos]   (V layout, pos contiguous)
// channel o maps to (d = o>>2, h = o&3) per reshape(B, DH, H, L).
// ---------------------------------------------------------------------------
__global__ __launch_bounds__(256) void proj_qkv_kernel(
    const float* __restrict__ W, const float* __restrict__ bias,
    const float* __restrict__ X, _Float16* __restrict__ out, int vmode) {
  __shared__ _Float16 ldsW[64 * 32];
  __shared__ _Float16 ldsX[128 * 32];
  const int t = threadIdx.x, L = t & 31, w = t >> 5;
  const int o0 = blockIdx.y * 64, n0 = blockIdx.x * 128, b = blockIdx.z;
  const int osub = (w & 3) * 16, nsub = (w >> 2) * 64;
  const float* Xb = X + (size_t)b * DIMc * Nc;

  v8f acc[4];
  v8f zz = {0, 0, 0, 0, 0, 0, 0, 0};
  acc[0] = zz; acc[1] = zz; acc[2] = zz; acc[3] = zz;

  for (int k0 = 0; k0 < DIMc; k0 += 32) {
    stage_w_tile(W, DIMc, k0, o0, t, ldsW);
    stage_x_f32(Xb, k0, n0, t, ldsX);
    __syncthreads();
    if (k0 + 32 < DIMc) {
      __builtin_prefetch(&W[(size_t)(k0 + 32) * DIMc + o0], 0, 1);
      __builtin_prefetch(&Xb[(size_t)(k0 + 32) * Nc + n0 + (t & 31) * 4], 0, 1);
    }
    wmma_tile_step(ldsW, ldsX, osub, nsub, L, acc);
    __syncthreads();
  }

  const int hi = L >> 4, col = L & 15;
  float bv[8];
#pragma unroll
  for (int j = 0; j < 8; ++j) bv[j] = bias[o0 + osub + j + 8 * hi];
#pragma unroll
  for (int f = 0; f < 4; ++f) {
#pragma unroll
    for (int j = 0; j < 8; ++j) {
      int o = o0 + osub + j + 8 * hi;
      int n = n0 + nsub + f * 16 + col;
      float v = acc[f][j] + bv[j];
      int d = o >> 2, hh = o & 3;
      size_t idx;
      if (vmode == 0)
        idx = (((size_t)b * Hc + hh) * Nc + n) * DHc + d;  // [b,h,pos,d]
      else
        idx = (((size_t)b * Hc + hh) * DHc + d) * Nc + n;  // [b,h,d,pos]
      out[idx] = (_Float16)v;
    }
  }
}

// ---------------------------------------------------------------------------
// Kernel 2: flash attention.  Block = 128 threads (4 waves); each wave owns a
// 16-row query tile.  K/V 64x64 tiles are fetched by the Tensor Data Mover
// into double-buffered LDS (fetch of tile i+1 overlaps WMMA work on tile i);
// wave 0 issues the TDM descriptors and gates on s_wait_tensorcnt.
// S = (Q/8) K (8 WMMAs/wave), online softmax via 16-lane shfl reductions,
// P round-trips through per-wave LDS to become A fragments, O += P V.
// ---------------------------------------------------------------------------
__global__ __launch_bounds__(128) void flash_attn_kernel(
    const _Float16* __restrict__ Qb, const _Float16* __restrict__ Kb,
    const _Float16* __restrict__ Vb, _Float16* __restrict__ attn) {
  __shared__ _Float16 ldsK[2][64 * 64];    // [m][d]
  __shared__ _Float16 ldsV[2][64 * 64];    // [d][m]
  __shared__ _Float16 ldsP[4 * 16 * 64];   // per-wave [n][m]
  const int t = threadIdx.x, L = t & 31, w = t >> 5;
  const int n0 = blockIdx.x * 64;
  const int h = blockIdx.y, b = blockIdx.z;
  const int hi = L >> 4, col = L & 15;

  // Q fragments (A-matrix, rows = query pos, K-dim = d), pre-scaled 1/sqrt(DH)
  const _Float16* Qrow =
      Qb + (((size_t)(b * Hc + h)) * Nc + (n0 + w * 16 + col)) * DHc;
  v16h qa0 = *(const v16h*)&Qrow[hi * 16];
  v16h qa1 = *(const v16h*)&Qrow[32 + hi * 16];
  qa0 *= (_Float16)0.125f;
  qa1 *= (_Float16)0.125f;

  const _Float16* Kbase = Kb + ((size_t)(b * Hc + h)) * Nc * DHc;  // [m][d]
  const _Float16* Vbase = Vb + ((size_t)(b * Hc + h)) * DHc * Nc;  // [d][m]

  v8f O[4];
  v8f zz = {0, 0, 0, 0, 0, 0, 0, 0};
  O[0] = zz; O[1] = zz; O[2] = zz; O[3] = zz;
  float mrun[8], lrun[8];
#pragma unroll
  for (int j = 0; j < 8; ++j) { mrun[j] = -1e30f; lrun[j] = 0.f; }

  // Prefetch tile 0 via TDM (wave 0 only; EXEC is ignored by tensor ops).
  if (w == 0) {
    tdm_load_2d_f16((unsigned)(uintptr_t)(void*)&ldsK[0][0], &Kbase[0],
                    /*tile*/ DHc, 64, /*tensor*/ DHc, Nc, /*stride*/ DHc);
    tdm_load_2d_f16((unsigned)(uintptr_t)(void*)&ldsV[0][0], &Vbase[0],
                    /*tile*/ 64, DHc, /*tensor*/ Nc, DHc, /*stride*/ Nc);
  }

  for (int m0 = 0; m0 < Nc; m0 += 64) {
    const int cur = (m0 >> 6) & 1;
    if (w == 0) __builtin_amdgcn_s_wait_tensorcnt(0);  // tile `cur` resident
    __syncthreads();  // all waves see tile; buffers from iter-1 fully read
    if (w == 0 && (m0 + 64) < Nc) {
      const int nxt = cur ^ 1;
      tdm_load_2d_f16((unsigned)(uintptr_t)(void*)&ldsK[nxt][0],
                      &Kbase[(size_t)(m0 + 64) * DHc], DHc, 64, DHc, Nc, DHc);
      tdm_load_2d_f16((unsigned)(uintptr_t)(void*)&ldsV[nxt][0],
                      &Vbase[m0 + 64], 64, DHc, Nc, DHc, Nc);
    }

    // S = Q K : 4 m-fragments x 2 d-chunks; fragments batch-loaded per chunk
    v8f S[4];
    S[0] = zz; S[1] = zz; S[2] = zz; S[3] = zz;
    {
      v16h bk[4];
#pragma unroll
      for (int f = 0; f < 4; ++f)
        bk[f] = *(const v16h*)&ldsK[cur][(f * 16 + col) * 64 + hi * 16];
#pragma unroll
      for (int f = 0; f < 4; ++f)
        S[f] = __builtin_amdgcn_wmma_f32_16x16x32_f16(false, qa0, false, bk[f],
                                                      (short)0, S[f], false,
                                                      false);
#pragma unroll
      for (int f = 0; f < 4; ++f)
        bk[f] = *(const v16h*)&ldsK[cur][(f * 16 + col) * 64 + 32 + hi * 16];
#pragma unroll
      for (int f = 0; f < 4; ++f)
        S[f] = __builtin_amdgcn_wmma_f32_16x16x32_f16(false, qa1, false, bk[f],
                                                      (short)0, S[f], false,
                                                      false);
    }

    // online softmax (row r = j + 8*hi lives across the 16-lane half)
    float sc[8];
#pragma unroll
    for (int j = 0; j < 8; ++j) {
      float mv = fmaxf(fmaxf(S[0][j], S[1][j]), fmaxf(S[2][j], S[3][j]));
      mv = red_max16(mv);
      float mn = fmaxf(mrun[j], mv);
      sc[j] = __expf(mrun[j] - mn);
      mrun[j] = mn;
    }
#pragma unroll
    for (int j = 0; j < 8; ++j) {
      float ls = 0.f;
#pragma unroll
      for (int f = 0; f < 4; ++f) {
        float p = __expf(S[f][j] - mrun[j]);
        S[f][j] = p;
        ls += p;
      }
      ls = red_sum16(ls);
      lrun[j] = lrun[j] * sc[j] + ls;
    }
#pragma unroll
    for (int f = 0; f < 4; ++f)
#pragma unroll
      for (int j = 0; j < 8; ++j) O[f][j] *= sc[j];

    // P : C-layout -> per-wave LDS -> A-layout (wave-internal, no barrier)
    _Float16* Pw = &ldsP[w * 16 * 64];
#pragma unroll
    for (int f = 0; f < 4; ++f)
#pragma unroll
      for (int j = 0; j < 8; ++j)
        Pw[(j + 8 * hi) * 64 + f * 16 + col] = (_Float16)S[f][j];

    v16h pa0 = *(const v16h*)&Pw[col * 64 + hi * 16];
    v16h pa1 = *(const v16h*)&Pw[col * 64 + 32 + hi * 16];
    {
      v16h bv[4];
#pragma unroll
      for (int f = 0; f < 4; ++f)
        bv[f] = *(const v16h*)&ldsV[cur][(f * 16 + col) * 64 + hi * 16];
#pragma unroll
      for (int f = 0; f < 4; ++f)
        O[f] = __builtin_amdgcn_wmma_f32_16x16x32_f16(false, pa0, false, bv[f],
                                                      (short)0, O[f], false,
                                                      false);
#pragma unroll
      for (int f = 0; f < 4; ++f)
        bv[f] = *(const v16h*)&ldsV[cur][(f * 16 + col) * 64 + 32 + hi * 16];
#pragma unroll
      for (int f = 0; f < 4; ++f)
        O[f] = __builtin_amdgcn_wmma_f32_16x16x32_f16(false, pa1, false, bv[f],
                                                      (short)0, O[f], false,
                                                      false);
    }
  }

#pragma unroll
  for (int f = 0; f < 4; ++f) {
#pragma unroll
    for (int j = 0; j < 8; ++j) {
      int d = f * 16 + col;
      int n = n0 + w * 16 + j + 8 * hi;
      float v = O[f][j] / lrun[j];
      attn[((size_t)b * DIMc + d * Hc + h) * Nc + n] = (_Float16)v;
    }
  }
}

// ---------------------------------------------------------------------------
// Kernel 3: message = Wo^T attn + bo   (f16 input, f16 output [b,c,n])
// ---------------------------------------------------------------------------
__global__ __launch_bounds__(256) void msg_gemm_kernel(
    const float* __restrict__ Wo, const float* __restrict__ bo,
    const _Float16* __restrict__ attn, _Float16* __restrict__ msg) {
  __shared__ _Float16 ldsW[64 * 32];
  __shared__ _Float16 ldsX[128 * 32];
  const int t = threadIdx.x, L = t & 31, w = t >> 5;
  const int o0 = blockIdx.y * 64, n0 = blockIdx.x * 128, b = blockIdx.z;
  const int osub = (w & 3) * 16, nsub = (w >> 2) * 64;
  const _Float16* Ab = attn + (size_t)b * DIMc * Nc;

  v8f acc[4];
  v8f zz = {0, 0, 0, 0, 0, 0, 0, 0};
  acc[0] = zz; acc[1] = zz; acc[2] = zz; acc[3] = zz;

  for (int k0 = 0; k0 < DIMc; k0 += 32) {
    stage_w_tile(Wo, DIMc, k0, o0, t, ldsW);
    stage_x_f16(Ab, k0, n0, t, ldsX);
    __syncthreads();
    if (k0 + 32 < DIMc)
      __builtin_prefetch(&Ab[(size_t)(k0 + 32) * Nc + n0 + (t & 31) * 4], 0, 1);
    wmma_tile_step(ldsW, ldsX, osub, nsub, L, acc);
    __syncthreads();
  }

  const int hi = L >> 4, col = L & 15;
  float bv[8];
#pragma unroll
  for (int j = 0; j < 8; ++j) bv[j] = bo[o0 + osub + j + 8 * hi];
#pragma unroll
  for (int f = 0; f < 4; ++f)
#pragma unroll
    for (int j = 0; j < 8; ++j) {
      int o = o0 + osub + j + 8 * hi;
      int n = n0 + nsub + f * 16 + col;
      msg[((size_t)b * DIMc + o) * Nc + n] = (_Float16)(acc[f][j] + bv[j]);
    }
}

// ---------------------------------------------------------------------------
// Kernel 4: h = W1^T [x ; msg] + b1  (K = 512 split 256/256), f32 out, plus
// per-channel sum / sum-of-squares atomics for train-mode BatchNorm.
// ---------------------------------------------------------------------------
__global__ __launch_bounds__(256) void h_gemm_kernel(
    const float* __restrict__ W1, const float* __restrict__ b1,
    const float* __restrict__ x, const _Float16* __restrict__ msg,
    float* __restrict__ hbuf, float* __restrict__ stats) {
  __shared__ _Float16 ldsW[64 * 32];
  __shared__ _Float16 ldsX[128 * 32];
  const int t = threadIdx.x, L = t & 31, w = t >> 5;
  const int o0 = blockIdx.y * 64, n0 = blockIdx.x * 128, b = blockIdx.z;
  const int osub = (w & 3) * 16, nsub = (w >> 2) * 64;
  const float* Xb = x + (size_t)b * DIMc * Nc;
  const _Float16* Mb = msg + (size_t)b * DIMc * Nc;

  v8f acc[4];
  v8f zz = {0, 0, 0, 0, 0, 0, 0, 0};
  acc[0] = zz; acc[1] = zz; acc[2] = zz; acc[3] = zz;

  for (int k0 = 0; k0 < DIM2; k0 += 32) {
    stage_w_tile(W1, DIM2, k0, o0, t, ldsW);
    if (k0 < DIMc)
      stage_x_f32(Xb, k0, n0, t, ldsX);
    else
      stage_x_f16(Mb, k0 - DIMc, n0, t, ldsX);
    __syncthreads();
    if (k0 + 32 < DIM2)
      __builtin_prefetch(&W1[(size_t)(k0 + 32) * DIM2 + o0], 0, 1);
    wmma_tile_step(ldsW, ldsX, osub, nsub, L, acc);
    __syncthreads();
  }

  const int hi = L >> 4, col = L & 15;
  float bv[8];
#pragma unroll
  for (int j = 0; j < 8; ++j) bv[j] = b1[o0 + osub + j + 8 * hi];
#pragma unroll
  for (int f = 0; f < 4; ++f)
#pragma unroll
    for (int j = 0; j < 8; ++j) acc[f][j] += bv[j];

  // store h (f32) and accumulate BN statistics
#pragma unroll
  for (int f = 0; f < 4; ++f)
#pragma unroll
    for (int j = 0; j < 8; ++j) {
      int o = o0 + osub + j + 8 * hi;
      int n = n0 + nsub + f * 16 + col;
      hbuf[((size_t)b * DIM2 + o) * Nc + n] = acc[f][j];
    }
#pragma unroll
  for (int j = 0; j < 8; ++j) {
    float s = 0.f, s2 = 0.f;
#pragma unroll
    for (int f = 0; f < 4; ++f) {
      float v = acc[f][j];
      s += v;
      s2 += v * v;
    }
    s = red_sum16(s);
    s2 = red_sum16(s2);
    if (col == 0) {
      int o = o0 + osub + j + 8 * hi;
      atomicAdd(&stats[o], s);
      atomicAdd(&stats[DIM2 + o], s2);
    }
  }
}

// ---------------------------------------------------------------------------
// Kernel 5: out = W2^T relu(bn(h)) + b2.  BN (mean/var from stats) + ReLU are
// fused into the B-operand staging (f32 -> f16 conversion).
// ---------------------------------------------------------------------------
__global__ __launch_bounds__(256) void out_gemm_kernel(
    const float* __restrict__ W2, const float* __restrict__ b2,
    const float* __restrict__ hbuf, const float* __restrict__ stats,
    const float* __restrict__ gamma, const float* __restrict__ beta,
    float* __restrict__ out) {
  __shared__ _Float16 ldsW[64 * 32];
  __shared__ _Float16 ldsX[128 * 32];
  const int t = threadIdx.x, L = t & 31, w = t >> 5;
  const int o0 = blockIdx.y * 64, n0 = blockIdx.x * 128, b = blockIdx.z;
  const int osub = (w & 3) * 16, nsub = (w >> 2) * 64;
  const float* Hb = hbuf + (size_t)b * DIM2 * Nc;
  const float invCnt = 1.0f / (float)(Bc * Nc);

  v8f acc[4];
  v8f zz = {0, 0, 0, 0, 0, 0, 0, 0};
  acc[0] = zz; acc[1] = zz; acc[2] = zz; acc[3] = zz;

  for (int k0 = 0; k0 < DIM2; k0 += 32) {
    stage_w_tile(W2, DIMc, k0, o0, t, ldsW);
    // stage h with fused BN + ReLU (batched loads, then cvt+store)
    float4 hv[4];
    float isg[4], sh[4];
#pragma unroll
    for (int i = 0; i < 4; ++i) {
      int lin = t + i * 256;
      int kk = lin >> 5, ng = (lin & 31) * 4;
      int k = k0 + kk;
      hv[i] = *(const float4*)&Hb[(size_t)k * Nc + n0 + ng];
      float mean = stats[k] * invCnt;
      float var = stats[DIM2 + k] * invCnt - mean * mean;
      isg[i] = rsqrtf(var + 1e-5f) * gamma[k];
      sh[i] = beta[k] - mean * isg[i];
    }
#pragma unroll
    for (int i = 0; i < 4; ++i) {
      int lin = t + i * 256;
      int kk = lin >> 5, ng = (lin & 31) * 4;
      ldsX[(ng + 0) * 32 + kk] = (_Float16)fmaxf(0.f, hv[i].x * isg[i] + sh[i]);
      ldsX[(ng + 1) * 32 + kk] = (_Float16)fmaxf(0.f, hv[i].y * isg[i] + sh[i]);
      ldsX[(ng + 2) * 32 + kk] = (_Float16)fmaxf(0.f, hv[i].z * isg[i] + sh[i]);
      ldsX[(ng + 3) * 32 + kk] = (_Float16)fmaxf(0.f, hv[i].w * isg[i] + sh[i]);
    }
    __syncthreads();
    if (k0 + 32 < DIM2)
      __builtin_prefetch(&Hb[(size_t)(k0 + 32) * Nc + n0 + (t & 31) * 4], 0, 1);
    wmma_tile_step(ldsW, ldsX, osub, nsub, L, acc);
    __syncthreads();
  }

  const int hi = L >> 4, col = L & 15;
  float bv[8];
#pragma unroll
  for (int j = 0; j < 8; ++j) bv[j] = b2[o0 + osub + j + 8 * hi];
#pragma unroll
  for (int f = 0; f < 4; ++f)
#pragma unroll
    for (int j = 0; j < 8; ++j) {
      int o = o0 + osub + j + 8 * hi;
      int n = n0 + nsub + f * 16 + col;
      out[((size_t)b * DIMc + o) * Nc + n] = acc[f][j] + bv[j];
    }
}

__global__ void zero_stats_kernel(float* stats) {
  int i = blockIdx.x * blockDim.x + threadIdx.x;
  if (i < 2 * DIM2) stats[i] = 0.f;
}

// ---------------------------------------------------------------------------
// Launch
// ---------------------------------------------------------------------------
extern "C" void kernel_launch(void* const* d_in, const int* in_sizes, int n_in,
                              void* d_out, int out_size, void* d_ws,
                              size_t ws_size, hipStream_t stream) {
  const float* x     = (const float*)d_in[0];
  const float* src   = (const float*)d_in[1];
  const float* Wq    = (const float*)d_in[2];
  const float* bq    = (const float*)d_in[3];
  const float* Wk    = (const float*)d_in[4];
  const float* bk    = (const float*)d_in[5];
  const float* Wv    = (const float*)d_in[6];
  const float* bv    = (const float*)d_in[7];
  const float* Wo    = (const float*)d_in[8];
  const float* bo    = (const float*)d_in[9];
  const float* W1    = (const float*)d_in[10];
  const float* b1    = (const float*)d_in[11];
  const float* gamma = (const float*)d_in[12];
  const float* beta  = (const float*)d_in[13];
  const float* W2    = (const float*)d_in[14];
  const float* b2    = (const float*)d_in[15];

  char* ws = (char*)d_ws;
  const size_t MB = 1024 * 1024;
  _Float16* Qb   = (_Float16*)(ws + 0 * MB);   // [B,H,N,DH] f16  (4 MiB)
  _Float16* Kb   = (_Float16*)(ws + 4 * MB);   // [B,H,M,DH] f16  (4 MiB)
  _Float16* Vb   = (_Float16*)(ws + 8 * MB);   // [B,H,DH,M] f16  (4 MiB)
  _Float16* attn = (_Float16*)(ws + 12 * MB);  // [B,DIM,N]  f16  (4 MiB)
  _Float16* msg  = (_Float16*)(ws + 16 * MB);  // [B,DIM,N]  f16  (4 MiB)
  float*    hbuf = (float*)   (ws + 20 * MB);  // [B,2DIM,N] f32  (16 MiB)
  float*    stats= (float*)   (ws + 36 * MB);  // 2*512 f32

  dim3 gProj(Nc / 128, DIMc / 64, Bc);
  proj_qkv_kernel<<<gProj, 256, 0, stream>>>(Wq, bq, x, Qb, 0);
  proj_qkv_kernel<<<gProj, 256, 0, stream>>>(Wk, bk, src, Kb, 0);
  proj_qkv_kernel<<<gProj, 256, 0, stream>>>(Wv, bv, src, Vb, 1);

  flash_attn_kernel<<<dim3(Nc / 64, Hc, Bc), 128, 0, stream>>>(Qb, Kb, Vb,
                                                               attn);

  msg_gemm_kernel<<<gProj, 256, 0, stream>>>(Wo, bo, attn, msg);

  zero_stats_kernel<<<4, 256, 0, stream>>>(stats);

  h_gemm_kernel<<<dim3(Nc / 128, DIM2 / 64, Bc), 256, 0, stream>>>(
      W1, b1, x, msg, hbuf, stats);

  out_gemm_kernel<<<dim3(Nc / 128, DIMc / 64, Bc), 256, 0, stream>>>(
      W2, b2, hbuf, stats, gamma, beta, (float*)d_out);
}
